// MSDeformAttn_71536975282230
// MI455X (gfx1250) — matrix operand order repaired and checked by reading, compiled, not run on hardware
//
#include <hip/hip_runtime.h>

// ---------------------------------------------------------------------------
// MSDeformAttn for MI455X (gfx1250, wave32, WMMA).
// - all GEMMs: V_WMMA_F32_16X16X32_BF16 (bf16 in, f32 accumulate)
// - operands pre-converted to bf16 once (halves GEMM-side HBM traffic)
// - tile staging via GLOBAL_LOAD_ASYNC_TO_LDS_B128 (ASYNCcnt path)
// ---------------------------------------------------------------------------

typedef __attribute__((ext_vector_type(16))) __bf16 v16bf;
typedef __attribute__((ext_vector_type(8)))  __bf16 v8bf;
typedef __attribute__((ext_vector_type(4)))  __bf16 v4bf;
typedef __attribute__((ext_vector_type(8)))  float  v8f;
typedef int v4i __attribute__((vector_size(16)));   // matches builtin param type

#define B_SZ 32
#define LQ_SZ 576
#define LV_SZ 3024
#define C_SZ 1024
#define KDIM 1024

// GEMM tiling
#define BM 128
#define BN 64
#define BK 64
#define LDA_S 72   // padded row stride (bf16 elements) for A tile (144 B)
#define LDB_S 72   // padded row stride for B tile

// ---------------------------------------------------------------------------
// Async global -> LDS copy (16 bytes per lane), gfx1250 ASYNCcnt path.
// ---------------------------------------------------------------------------
static __device__ __forceinline__ void async_g2l_b128(const void* gsrc, void* ldst)
{
#if __has_builtin(__builtin_amdgcn_global_load_async_to_lds_b128)
    // prototype (from clang diagnostic): (global v4i*, local v4i*, int, int)
    __builtin_amdgcn_global_load_async_to_lds_b128(
        (__attribute__((address_space(1))) v4i*)(unsigned long long)gsrc,
        (__attribute__((address_space(3))) v4i*)(unsigned int)(unsigned long long)ldst,
        0, 0);
#else
    // ISA: VDST = LDS byte address VGPR, ADDR = 64-bit global address (GV mode).
    unsigned int l32 = (unsigned int)(unsigned long long)ldst; // LDS offset = addr[31:0]
    asm volatile("global_load_async_to_lds_b128 %0, %1, off"
                 :: "v"(l32), "v"(gsrc) : "memory");
#endif
}

static __device__ __forceinline__ void wait_asynccnt0()
{
#if __has_builtin(__builtin_amdgcn_s_wait_asynccnt)
    __builtin_amdgcn_s_wait_asynccnt(0);
#else
    asm volatile("s_wait_asynccnt 0x0" ::: "memory");
#endif
}

// ---------------------------------------------------------------------------
// f32 -> bf16 elementwise conversion. n must be a multiple of 1024.
// Grid: n/1024 blocks, 256 threads; 4 elements per thread.
// ---------------------------------------------------------------------------
__global__ __launch_bounds__(256) void f32_to_bf16_kernel(
    const float* __restrict__ src, __bf16* __restrict__ dst)
{
    const long i = ((long)blockIdx.x * 256 + threadIdx.x) * 4;
    const float4 f = *(const float4*)&src[i];
    v4bf o;
    o[0] = (__bf16)f.x; o[1] = (__bf16)f.y;
    o[2] = (__bf16)f.z; o[3] = (__bf16)f.w;
    *(v4bf*)&dst[i] = o;
}

// ---------------------------------------------------------------------------
// Generic GEMM: out[m, colOff + n] = sum_k A[m,k] * W[n,k] + bias[n]
// A: M x K row-major bf16 (K = 1024), W: N x K row-major bf16.
// Grid: (N/BN, M/BM), block 256 threads (8 waves).
// ---------------------------------------------------------------------------
template <bool OUT_BF16>
__global__ __launch_bounds__(256) void gemm_wmma(
    const __bf16* __restrict__ A, const __bf16* __restrict__ W,
    const float* __restrict__ bias, void* __restrict__ outv,
    int ldo, int colOff)
{
    __shared__ __bf16 As[BM * LDA_S];
    __shared__ __bf16 Bs[BN * LDB_S];

    const int tid   = threadIdx.x;
    const int wave  = tid >> 5;
    const int lane  = tid & 31;
    const int lhalf = lane >> 4;     // 0: lanes 0-15, 1: lanes 16-31
    const int l16   = lane & 15;

    const long mBase = (long)blockIdx.y * BM;
    const int  nBase = blockIdx.x * BN;

    v8f acc[4];
    #pragma unroll
    for (int nt = 0; nt < 4; ++nt) {
        const float bb = bias[nBase + nt * 16 + l16];
        #pragma unroll
        for (int e = 0; e < 8; ++e) acc[nt][e] = bb;
    }

    // staging map: 16-byte chunks; chunk = 8 bf16 along K, rows advance by 32
    const int chunk = tid & 7;       // 0..7  -> k offset chunk*8
    const int rowId = tid >> 3;      // 0..31

    const int mRow = wave * 16 + l16;

    for (int k0 = 0; k0 < KDIM; k0 += BK) {
        // ---- async stage A tile: BM x BK bf16 ----
        #pragma unroll
        for (int r = 0; r < 4; ++r) {
            const int m = rowId + 32 * r;                    // 0..127
            async_g2l_b128(&A[(mBase + m) * KDIM + k0 + chunk * 8],
                           &As[m * LDA_S + chunk * 8]);
        }
        // ---- async stage B tile: BN x BK bf16 from W ----
        #pragma unroll
        for (int r = 0; r < 2; ++r) {
            const int n = rowId + 32 * r;                    // 0..63
            async_g2l_b128(&W[(long)(nBase + n) * KDIM + k0 + chunk * 8],
                           &Bs[n * LDB_S + chunk * 8]);
        }
        wait_asynccnt0();
        __syncthreads();

        // ---- preload ALL fragments for this stage, then 8 back-to-back WMMAs
        v16bf af[2];
        v16bf bfr[2][4];
        #pragma unroll
        for (int ks2 = 0; ks2 < 2; ++ks2) {
            const int ks = ks2 * 32;
            // A fragment: lanes 0-15 hold K = ks+0..7 & ks+16..23 of row m,
            //             lanes 16-31 hold K = ks+8..15 & ks+24..31.
            const v8bf alo = *(const v8bf*)&As[mRow * LDA_S + ks + lhalf * 8];
            const v8bf ahi = *(const v8bf*)&As[mRow * LDA_S + ks + lhalf * 8 + 16];
            af[ks2] = __builtin_shufflevector(alo, ahi,
                0,1,2,3,4,5,6,7,8,9,10,11,12,13,14,15);
            #pragma unroll
            for (int nt = 0; nt < 4; ++nt) {
                const int n = nt * 16 + l16;
                // B fragment: lanes 0-15 hold K = ks+0..7 & ks+16..23 of column n,
                //             lanes 16-31 hold K = ks+8..15 & ks+24..31.
                const v8bf blo = *(const v8bf*)&Bs[n * LDB_S + ks + lhalf * 8];
                const v8bf bhi = *(const v8bf*)&Bs[n * LDB_S + ks + lhalf * 8 + 16];
                bfr[ks2][nt] = __builtin_shufflevector(blo, bhi,
                    0,1,2,3,4,5,6,7,8,9,10,11,12,13,14,15);
            }
        }
        #pragma unroll
        for (int ks2 = 0; ks2 < 2; ++ks2) {
            #pragma unroll
            for (int nt = 0; nt < 4; ++nt) {
                acc[nt] = __builtin_amdgcn_wmma_f32_16x16x32_bf16(
                    /*neg_a=*/false, af[ks2], /*neg_b=*/false, bfr[ks2][nt],
                    /*c_mod=*/(short)0, acc[nt],
                    /*reuse_a=*/false, /*reuse_b=*/false);
            }
        }
        __syncthreads();
    }

    // ---- store C: VGPR r -> (M = r + 8*lhalf, N = l16) within wave's tile ----
    #pragma unroll
    for (int nt = 0; nt < 4; ++nt) {
        const int col = colOff + nBase + nt * 16 + l16;
        #pragma unroll
        for (int r = 0; r < 8; ++r) {
            const long row = mBase + wave * 16 + lhalf * 8 + r;
            if constexpr (OUT_BF16) {
                ((__bf16*)outv)[row * ldo + col] = (__bf16)acc[nt][r];
            } else {
                ((float*)outv)[row * ldo + col] = acc[nt][r];
            }
        }
    }
}

// ---------------------------------------------------------------------------
// Reference-point prep: ref = (mean48x48 + direct24x24 + bilin(12x12->24x24))/3
// Grid: B blocks, 256 threads.
// ---------------------------------------------------------------------------
__global__ __launch_bounds__(256) void ref_points_kernel(
    const float* __restrict__ rp, float* __restrict__ ref)
{
    const int b = blockIdx.x;
    const float* r = &rp[(long)b * LV_SZ * 2];
    __shared__ float sx[256], sy[256];

    float ax = 0.f, ay = 0.f;
    for (int i = threadIdx.x; i < 2304; i += 256) {
        ax += r[i * 2 + 0];
        ay += r[i * 2 + 1];
    }
    sx[threadIdx.x] = ax; sy[threadIdx.x] = ay;
    __syncthreads();
    for (int s = 128; s > 0; s >>= 1) {
        if (threadIdx.x < s) {
            sx[threadIdx.x] += sx[threadIdx.x + s];
            sy[threadIdx.x] += sy[threadIdx.x + s];
        }
        __syncthreads();
    }
    const float p1x = sx[0] * (1.0f / 2304.f);
    const float p1y = sy[0] * (1.0f / 2304.f);

    const float* g = &r[2880 * 2];   // 12x12 level
    for (int q = threadIdx.x; q < LQ_SZ; q += 256) {
        const int rr = q / 24, cc = q % 24;
        const float p2x = r[(2304 + q) * 2 + 0];
        const float p2y = r[(2304 + q) * 2 + 1];

        const float fr = rr * 0.5f - 0.25f;
        const float fc = cc * 0.5f - 0.25f;
        int i0 = (int)floorf(fr); const float wa = fr - (float)i0;
        int j0 = (int)floorf(fc); const float wb = fc - (float)j0;
        int i1 = i0 + 1, j1 = j0 + 1;
        i0 = min(max(i0, 0), 11); i1 = min(max(i1, 0), 11);
        j0 = min(max(j0, 0), 11); j1 = min(max(j1, 0), 11);

        float p3x, p3y;
        {
            const float v00 = g[(i0 * 12 + j0) * 2], v01 = g[(i0 * 12 + j1) * 2];
            const float v10 = g[(i1 * 12 + j0) * 2], v11 = g[(i1 * 12 + j1) * 2];
            p3x = (1.f - wa) * ((1.f - wb) * v00 + wb * v01)
                +        wa  * ((1.f - wb) * v10 + wb * v11);
        }
        {
            const float v00 = g[(i0 * 12 + j0) * 2 + 1], v01 = g[(i0 * 12 + j1) * 2 + 1];
            const float v10 = g[(i1 * 12 + j0) * 2 + 1], v11 = g[(i1 * 12 + j1) * 2 + 1];
            p3y = (1.f - wa) * ((1.f - wb) * v00 + wb * v01)
                +        wa  * ((1.f - wb) * v10 + wb * v11);
        }
        ref[((long)b * LQ_SZ + q) * 2 + 0] = (p1x + p2x + p3x) * (1.f / 3.f);
        ref[((long)b * LQ_SZ + q) * 2 + 1] = (p1y + p2y + p3y) * (1.f / 3.f);
    }
}

// ---------------------------------------------------------------------------
// Sampling: per (b,q): softmax over 12 logits per head, nearest-cell gather of
// bf16 v, f32 weighted accumulate, bf16 output (feeds final WMMA GEMM).
// Thread t -> head = t>>4, 4-elem chunk = (t&15)*4. Grid: B*LQ blocks.
// ---------------------------------------------------------------------------
__global__ __launch_bounds__(256) void msda_sample_kernel(
    const __bf16* __restrict__ v, const float* __restrict__ qp,
    const float* __restrict__ ref, __bf16* __restrict__ tmp)
{
    const int bq = blockIdx.x;
    const int b  = bq / LQ_SZ;
    const int t  = threadIdx.x;
    const int h  = t >> 4;
    const int i  = t & 15;

    const float rx = ref[(long)bq * 2 + 0];
    const float ry = ref[(long)bq * 2 + 1];
    const float* q = &qp[(long)bq * 576];

    // softmax over 12 (NL*NP) attention logits for this head
    float lg[12];
    float mx = -1e30f;
    #pragma unroll
    for (int j = 0; j < 12; ++j) {
        lg[j] = q[384 + h * 12 + j];
        mx = fmaxf(mx, lg[j]);
    }
    float den = 0.f;
    #pragma unroll
    for (int j = 0; j < 12; ++j) { lg[j] = __expf(lg[j] - mx); den += lg[j]; }
    const float inv = 1.0f / den;

    float4 acc = make_float4(0.f, 0.f, 0.f, 0.f);
    const int Hs[3] = {48, 24, 12};
    const int Ss[3] = {0, 2304, 2880};
    #pragma unroll
    for (int lvl = 0; lvl < 3; ++lvl) {
        const int HW = Hs[lvl];
        #pragma unroll
        for (int p = 0; p < 4; ++p) {
            const int idx = ((h * 3 + lvl) * 4 + p);
            const float cx = q[idx * 2 + 0] + rx;
            const float cy = q[idx * 2 + 1] + ry;
            int xi = (int)floorf(cx); xi = min(max(xi, 0), HW - 1);
            int yi = (int)floorf(cy); yi = min(max(yi, 0), HW - 1);
            const long pos = (long)b * LV_SZ + Ss[lvl] + xi * HW + yi;
            const v4bf gv = *(const v4bf*)&v[pos * C_SZ + h * 64 + i * 4];
            const float w = lg[lvl * 4 + p] * inv;
            acc.x += w * (float)gv[0]; acc.y += w * (float)gv[1];
            acc.z += w * (float)gv[2]; acc.w += w * (float)gv[3];
        }
    }
    v4bf o;
    o[0] = (__bf16)acc.x; o[1] = (__bf16)acc.y;
    o[2] = (__bf16)acc.z; o[3] = (__bf16)acc.w;
    *(v4bf*)&tmp[(long)bq * C_SZ + h * 64 + i * 4] = o;
}

// ---------------------------------------------------------------------------
// Launch
// ---------------------------------------------------------------------------
extern "C" void kernel_launch(void* const* d_in, const int* in_sizes, int n_in,
                              void* d_out, int out_size, void* d_ws, size_t ws_size,
                              hipStream_t stream)
{
    const float* query = (const float*)d_in[0];
    const float* rp    = (const float*)d_in[1];
    const float* value = (const float*)d_in[2];
    // d_in[3] = spatial_shapes, d_in[4] = level_start_index (compile-time here)
    const float* Wv    = (const float*)d_in[5];
    const float* bv    = (const float*)d_in[6];
    const float* Wo    = (const float*)d_in[7];
    const float* bo    = (const float*)d_in[8];
    const float* Ww    = (const float*)d_in[9];
    const float* bw    = (const float*)d_in[10];
    const float* Wout  = (const float*)d_in[11];
    const float* bout  = (const float*)d_in[12];
    float* out = (float*)d_out;

    const size_t nBLV = (size_t)B_SZ * LV_SZ;          // 96768
    const size_t nBLQ = (size_t)B_SZ * LQ_SZ;          // 18432

    // workspace carve-up (all chunks are multiples of 16 bytes)
    char* p = (char*)d_ws;
    __bf16* vbf   = (__bf16*)p; p += nBLV * C_SZ * 2;          // bf16 value
    __bf16* vproj = (__bf16*)p; p += nBLV * C_SZ * 2;          // bf16 projected v
    __bf16* qbf   = (__bf16*)p; p += nBLQ * C_SZ * 2;          // bf16 query
    __bf16* Wvb   = (__bf16*)p; p += (size_t)C_SZ * C_SZ * 2;
    __bf16* Wob   = (__bf16*)p; p += (size_t)384 * C_SZ * 2;
    __bf16* Wwb   = (__bf16*)p; p += (size_t)192 * C_SZ * 2;
    __bf16* Woutb = (__bf16*)p; p += (size_t)C_SZ * C_SZ * 2;
    __bf16* tmpbf = (__bf16*)p; p += nBLQ * C_SZ * 2;          // sampled output
    float*  qp    = (float*)p;  p += nBLQ * 576 * 4;           // offsets+logits
    float*  refb  = (float*)p;  p += nBLQ * 2 * 4;

    // 0. one-time bf16 conversions (n/1024 blocks each)
    f32_to_bf16_kernel<<<(int)(nBLV * C_SZ / 1024), 256, 0, stream>>>(value, vbf);
    f32_to_bf16_kernel<<<(int)(nBLQ * C_SZ / 1024), 256, 0, stream>>>(query, qbf);
    f32_to_bf16_kernel<<<(int)(C_SZ * C_SZ / 1024), 256, 0, stream>>>(Wv, Wvb);
    f32_to_bf16_kernel<<<(int)(384 * C_SZ / 1024),  256, 0, stream>>>(Wo, Wob);
    f32_to_bf16_kernel<<<(int)(192 * C_SZ / 1024),  256, 0, stream>>>(Ww, Wwb);
    f32_to_bf16_kernel<<<(int)(C_SZ * C_SZ / 1024), 256, 0, stream>>>(Wout, Woutb);

    // 1. reference points
    ref_points_kernel<<<B_SZ, 256, 0, stream>>>(rp, refb);

    // 2. value projection: vproj = value @ Wv^T + bv (M=96768, N=1024), bf16 out
    gemm_wmma<true><<<dim3(C_SZ / BN, (int)(nBLV / BM)), 256, 0, stream>>>(
        vbf, Wvb, bv, vproj, C_SZ, 0);

    // 3. query projections into qp[:, 0:384] (offsets) and qp[:, 384:576] (logits)
    gemm_wmma<false><<<dim3(384 / BN, (int)(nBLQ / BM)), 256, 0, stream>>>(
        qbf, Wob, bo, qp, 576, 0);
    gemm_wmma<false><<<dim3(192 / BN, (int)(nBLQ / BM)), 256, 0, stream>>>(
        qbf, Wwb, bw, qp, 576, 384);

    // 4. deformable sampling + softmax-weighted accumulation (bf16 out)
    msda_sample_kernel<<<(int)nBLQ, 256, 0, stream>>>(vproj, qp, refb, tmpbf);

    // 5. output projection: out = tmp @ Wout^T + bout (M=18432, N=1024), f32 out
    gemm_wmma<false><<<dim3(C_SZ / BN, (int)(nBLQ / BM)), 256, 0, stream>>>(
        tmpbf, Woutb, bout, out, C_SZ, 0);
}
